// Attention_970662609306
// MI455X (gfx1250) — matrix operand order repaired and checked by reading, not compiled, not run on hardware
//
#include <hip/hip_runtime.h>

typedef __attribute__((ext_vector_type(16))) _Float16 v16h;
typedef __attribute__((ext_vector_type(8)))  float    v8f;

constexpr int S = 4096, H = 16, D = 64;
constexpr int HD = H * D;          // 1024 floats per (seq) row
constexpr int KV_TILE = 128;       // keys per inner iteration
constexpr int WAVES = 8;           // 8 waves -> 128 query rows per block
// Q pre-scale: (1/sqrt(64)) * log2(e)  -> softmax runs in exp2 domain
#define QSCALE 0.1803368801111244f

__device__ inline unsigned pk2(float a, float b) {
  _Float16 x = (_Float16)a, y = (_Float16)b;
  unsigned short ux = __builtin_bit_cast(unsigned short, x);
  unsigned short uy = __builtin_bit_cast(unsigned short, y);
  return (unsigned)ux | ((unsigned)uy << 16);
}

__global__ __launch_bounds__(256) void fa_fwd_kernel(
    const float* __restrict__ Qg, const float* __restrict__ Kg,
    const float* __restrict__ Vg, float* __restrict__ Og)
{
  __shared__ alignas(32) _Float16 Kl[KV_TILE * D];         // [key][d]  16 KB
  __shared__ alignas(32) _Float16 Vt[D * KV_TILE];         // [d][key]  16 KB
  __shared__ alignas(32) float    Sl[WAVES][16 * KV_TILE]; // scores    64 KB
  __shared__ alignas(32) float    Al[WAVES][16];           // alpha     512 B

  const int tid  = threadIdx.x;
  const int lane = tid & 31;
  const int w    = tid >> 5;
  const int n    = lane & 15;      // B/C column, and A-format row
  const int half = lane >> 4;
  const int h    = blockIdx.y;
  const int qbase = blockIdx.x * (16 * WAVES) + w * 16;

  // ---- Q tile in A-format, pre-scaled by log2(e)/sqrt(D) ----
  const float* qrow = Qg + (size_t)(qbase + n) * HD + h * D;
  v16h qa[2];
  #pragma unroll
  for (int ah = 0; ah < 2; ++ah) {
    const float* p0 = qrow + 32 * ah + 16 * half;
    const float* p1 = qrow + 32 * ah + 16 * half + 8;
    v16h v;
    #pragma unroll
    for (int i = 0; i < 8; ++i) v[i]     = (_Float16)(p0[i] * QSCALE);
    #pragma unroll
    for (int i = 0; i < 8; ++i) v[8 + i] = (_Float16)(p1[i] * QSCALE);
    qa[ah] = v;
  }

  v16h ones;                           // all-ones B operand: rowsum via WMMA
  #pragma unroll
  for (int i = 0; i < 16; ++i) ones[i] = (_Float16)1.0f;

  float mrow = -__builtin_inff();      // running max (log2 domain), row m = n
  float lrow[8];                       // running sums, C-layout rows r+8*half
  v8f o[4] = {};
  #pragma unroll
  for (int r = 0; r < 8; ++r) lrow[r] = 0.f;

  // loader mappings: K: 1 key x 32 d per thread; V: 2 adjacent keys x 16 d
  const int kkey = tid >> 1, kchunk = (tid & 1) * 32;
  const int vkey = (tid >> 2) << 1, vd = (tid & 3) * 16;

  for (int kb = 0; kb < S; kb += KV_TILE) {
    __syncthreads();               // previous tile fully consumed
    // ---- cooperative K/V tile load, fp32 -> fp16 ----
    {
      const float* kr = Kg + (size_t)(kb + kkey) * HD + h * D + kchunk;
      float4 kq[8];
      #pragma unroll
      for (int i = 0; i < 8; ++i) kq[i] = ((const float4*)kr)[i];
      uint4* kdst = (uint4*)&Kl[kkey * D + kchunk];
      #pragma unroll
      for (int i = 0; i < 4; ++i)
        kdst[i] = make_uint4(pk2(kq[2*i].x,   kq[2*i].y),
                             pk2(kq[2*i].z,   kq[2*i].w),
                             pk2(kq[2*i+1].x, kq[2*i+1].y),
                             pk2(kq[2*i+1].z, kq[2*i+1].w));
      const float* vr = Vg + (size_t)(kb + vkey) * HD + h * D + vd;
      float4 va[4], vb4[4];
      #pragma unroll
      for (int i = 0; i < 4; ++i) { va[i]  = ((const float4*)vr)[i & 3];
                                    va[i]  = ((const float4*)vr)[i];
                                    vb4[i] = ((const float4*)(vr + HD))[i]; }
      float af[16], bf[16];
      #pragma unroll
      for (int i = 0; i < 4; ++i) {
        af[4*i+0]=va[i].x;  af[4*i+1]=va[i].y;  af[4*i+2]=va[i].z;  af[4*i+3]=va[i].w;
        bf[4*i+0]=vb4[i].x; bf[4*i+1]=vb4[i].y; bf[4*i+2]=vb4[i].z; bf[4*i+3]=vb4[i].w;
      }
      #pragma unroll
      for (int i = 0; i < 16; ++i)   // two adjacent keys -> one packed dword
        *(unsigned*)&Vt[(vd + i) * KV_TILE + vkey] = pk2(af[i], bf[i]);
      if (kb + KV_TILE < S) {
        __builtin_prefetch(kr + (size_t)KV_TILE * HD, 0, 1);
        __builtin_prefetch(vr + (size_t)KV_TILE * HD, 0, 1);
      }
    }
    __syncthreads();               // LDS tile ready

    // ---- scores = (Q') * K^T : eight 16x16 C tiles -> row-major LDS ----
    float* Sw = &Sl[w][0];
    #pragma unroll
    for (int g = 0; g < 8; ++g) {
      const v16h kb0 = *(const v16h*)&Kl[(16 * g + n) * D + 0  + 16 * half];
      const v16h kb1 = *(const v16h*)&Kl[(16 * g + n) * D + 32 + 16 * half];
      v8f acc = {};
      acc = __builtin_amdgcn_wmma_f32_16x16x32_f16(false, qa[0], false, kb0,
                                                   (short)0, acc, false, false);
      acc = __builtin_amdgcn_wmma_f32_16x16x32_f16(false, qa[1], false, kb1,
                                                   (short)0, acc, false, false);
      #pragma unroll
      for (int r = 0; r < 8; ++r)
        Sw[(r + 8 * half) * KV_TILE + 16 * g + n] = acc[r];
    }
    asm volatile("s_wait_dscnt 0" ::: "memory");
    v8f sp[8];                      // this lane's 64 A-format score elements
    #pragma unroll
    for (int j = 0; j < 8; ++j)
      sp[j] = *(const v8f*)&Sw[n * KV_TILE + 32 * (j >> 1) + 16 * half + 8 * (j & 1)];

    // ---- in-lane max tree + one half-combine shuffle ----
    float x[64];
    #pragma unroll
    for (int j = 0; j < 8; ++j)
      #pragma unroll
      for (int i = 0; i < 8; ++i) x[8 * j + i] = sp[j][i];
    #pragma unroll
    for (int st = 32; st >= 1; st >>= 1)
      #pragma unroll
      for (int i = 0; i < st; ++i) x[i] = fmaxf(x[i], x[i + st]);
    const float rowmax = fmaxf(x[0], __shfl_xor(x[0], 16, 32));

    const float mnew  = fmaxf(mrow, rowmax);
    const float alpha = __builtin_amdgcn_exp2f(mrow - mnew);
    mrow = mnew;

    // ---- P = exp2(S' - m') directly into A-format f16 (four A tiles) ----
    v16h pa[4];
    #pragma unroll
    for (int t = 0; t < 4; ++t)
      #pragma unroll
      for (int i = 0; i < 8; ++i) {
        pa[t][i]     = (_Float16)__builtin_amdgcn_exp2f(sp[2*t][i]   - mnew);
        pa[t][8 + i] = (_Float16)__builtin_amdgcn_exp2f(sp[2*t+1][i] - mnew);
      }

    // ---- alpha: row-layout -> C-layout via tiny LDS broadcast ----
    if (half == 0) Al[w][n] = alpha;
    asm volatile("s_wait_dscnt 0" ::: "memory");
    const v8f ac = *(const v8f*)&Al[w][8 * half];

    // ---- rowsum(P) via WMMA against all-ones B ----
    v8f rs = {};
    #pragma unroll
    for (int t = 0; t < 4; ++t)
      rs = __builtin_amdgcn_wmma_f32_16x16x32_f16(false, pa[t], false, ones,
                                                  (short)0, rs, false, false);

    // ---- rescale O, O += P*V (4 WMMAs per c-tile), update l ----
    #pragma unroll
    for (int c = 0; c < 4; ++c)
      #pragma unroll
      for (int r = 0; r < 8; ++r) o[c][r] *= ac[r];
    #pragma unroll
    for (int c = 0; c < 4; ++c)
      #pragma unroll
      for (int kg = 0; kg < 4; ++kg) {
        const v16h vb = *(const v16h*)
            &Vt[(16 * c + n) * KV_TILE + 32 * kg + 16 * half];
        o[c] = __builtin_amdgcn_wmma_f32_16x16x32_f16(false, pa[kg], false, vb,
                                                      (short)0, o[c], false, false);
      }
    #pragma unroll
    for (int r = 0; r < 8; ++r) lrow[r] = lrow[r] * ac[r] + rs[r];
  }

  // ---- epilogue: O / l ----
  #pragma unroll
  for (int r = 0; r < 8; ++r) {
    const float inv = 1.f / lrow[r];
    const int q = qbase + r + 8 * half;
    float* orow = Og + (size_t)q * HD + h * D;
    #pragma unroll
    for (int c = 0; c < 4; ++c) orow[16 * c + n] = o[c][r] * inv;
  }
}

extern "C" void kernel_launch(void* const* d_in, const int* in_sizes, int n_in,
                              void* d_out, int out_size, void* d_ws, size_t ws_size,
                              hipStream_t stream) {
  const float* q = (const float*)d_in[0];
  const float* k = (const float*)d_in[1];
  const float* v = (const float*)d_in[2];
  float* out = (float*)d_out;
  dim3 grid(S / (16 * WAVES), H);   // (32, 16)
  dim3 block(32 * WAVES);           // 256 threads = 8 waves
  hipLaunchKernelGGL(fa_fwd_kernel, grid, block, 0, stream, q, k, v, out);
}
